// GQABlock_17566416240829
// MI455X (gfx1250) — matrix operand order repaired and checked by reading, not compiled, not run on hardware
//
#include <hip/hip_runtime.h>

// ---------------------------------------------------------------------------
// GQA block for MI455X (gfx1250, wave32, WMMA 16x16x32 f16 -> f32 accum).
//   1) gemm_f16: Q/K/V projections and final O @ Wo (all K = 2048)
//      - double-buffered LDS, async global->LDS copy for the A tile
//        (GLOBAL_LOAD_ASYNC_TO_LDS_B128 / ASYNCcnt) when the builtin exists,
//        software-pipelined register prefetch + transpose for the B tile,
//      - LDS-bounced epilogue for coalesced 16B stores.
//   2) attn: flash attention, 4 waves/block, 16 q-rows per wave.
// Workspace: Q 16MB | K 4MB | V 4MB | O 16MB  (f16), total 40 MB.
// ---------------------------------------------------------------------------

typedef _Float16 half_t;
typedef __attribute__((ext_vector_type(16))) _Float16 v16h;
typedef __attribute__((ext_vector_type(8)))  _Float16 v8h;
typedef __attribute__((ext_vector_type(8)))  float    v8f;
typedef __attribute__((ext_vector_type(4)))  int      v4i;

// Builtin signature (from hipcc diagnostic): param1 = AS(1) v4i* (global),
// param2 = AS(3) v4i* (LDS), then two constant ints (offset, cpol).
typedef __attribute__((address_space(1))) v4i* gptr_b128;
typedef __attribute__((address_space(3))) v4i* lptr_b128;

#if defined(__has_builtin)
#if __has_builtin(__builtin_amdgcn_global_load_async_to_lds_b128) && \
    __has_builtin(__builtin_amdgcn_s_wait_asynccnt)
#define HAVE_ASYNC_LDS 1
#endif
#endif

__device__ __forceinline__ v16h cat16(v8h lo, v8h hi) {
    return __builtin_shufflevector(lo, hi, 0,1,2,3,4,5,6,7,8,9,10,11,12,13,14,15);
}

__device__ __forceinline__ v8f wmma_f16(v16h a, v16h b, v8f c) {
    // emits v_wmma_f32_16x16x32_f16
    return __builtin_amdgcn_wmma_f32_16x16x32_f16(false, a, false, b,
                                                  (short)0, c, false, false);
}

// ===========================================================================
// Tiled f16 GEMM: C[M,N] = A[M,K] @ B[K,N], f32 accumulate, f16 store.
// 256 threads = 8 waves (2x4). Block tile 128x128, K step 32, double buffer.
// ===========================================================================
__global__ __launch_bounds__(256)
void gemm_f16_kernel(const half_t* __restrict__ A,
                     const half_t* __restrict__ B,
                     half_t* __restrict__ C,
                     int M, int N, int K) {
    constexpr int BM = 128, BN = 128, BK = 32, LD = BK + 8;  // 80B row stride

    // 2x A tile + 2x B(transposed) tile = 40960 B; epilogue tile aliases it.
    __shared__ __align__(16) char smem_raw[4 * 128 * LD * (int)sizeof(half_t)];
    half_t (*As)[BM][LD] = reinterpret_cast<half_t (*)[BM][LD]>(smem_raw);
    half_t (*Bt)[BN][LD] =
        reinterpret_cast<half_t (*)[BN][LD]>(smem_raw + 2 * BM * LD * sizeof(half_t));

    const int tid   = threadIdx.x;
    const int lane  = tid & 31;
    const int wave  = tid >> 5;
    const int l16   = lane & 15;
    const int lhalf = lane >> 4;          // 0 / 1
    const int m0    = blockIdx.y * BM;
    const int n0    = blockIdx.x * BN;
    const int wm    = (wave >> 2) * 64;   // wave M offset: 0 / 64
    const int wn    = (wave & 3)  * 32;   // wave N offset: 0..96

    v8f acc[4][2] = {};                   // 64x32 of C per wave

    // --- staging helpers ----------------------------------------------------
#ifdef HAVE_ASYNC_LDS
    auto stageA_async = [&](int k0, int buf) {
        #pragma unroll
        for (int p = 0; p < 2; ++p) {
            int idx = tid + p * 256;            // 0..511
            int r   = idx >> 2;                 // 0..127
            int c   = (idx & 3) * 8;            // 0,8,16,24
            __builtin_amdgcn_global_load_async_to_lds_b128(
                (gptr_b128)(A + (size_t)(m0 + r) * K + k0 + c),
                (lptr_b128)&As[buf][r][c], 0, 0);
        }
    };
#else
    v8h areg[2];
    auto loadA = [&](int k0) {
        #pragma unroll
        for (int p = 0; p < 2; ++p) {
            int idx = tid + p * 256;
            int r   = idx >> 2;
            int c   = (idx & 3) * 8;
            areg[p] = *(const v8h*)(A + (size_t)(m0 + r) * K + k0 + c);
        }
    };
    auto storeA = [&](int buf) {
        #pragma unroll
        for (int p = 0; p < 2; ++p) {
            int idx = tid + p * 256;
            int r   = idx >> 2;
            int c   = (idx & 3) * 8;
            *(v8h*)(&As[buf][r][c]) = areg[p];
        }
    };
#endif
    v8h breg[2];
    auto loadB = [&](int k0) {
        #pragma unroll
        for (int p = 0; p < 2; ++p) {
            int idx = tid + p * 256;            // 0..511
            int kk  = idx >> 4;                 // 0..31
            int c   = (idx & 15) * 8;           // 0..120
            breg[p] = *(const v8h*)(B + (size_t)(k0 + kk) * N + n0 + c);
        }
    };
    auto storeBt = [&](int buf) {               // transpose scatter
        #pragma unroll
        for (int p = 0; p < 2; ++p) {
            int idx = tid + p * 256;
            int kk  = idx >> 4;
            int c   = (idx & 15) * 8;
            #pragma unroll
            for (int j = 0; j < 8; ++j) Bt[buf][c + j][kk] = breg[p][j];
        }
    };

    // --- prologue: stage k-step 0 into buffer 0 -----------------------------
#ifdef HAVE_ASYNC_LDS
    stageA_async(0, 0);
#else
    loadA(0); storeA(0);
#endif
    loadB(0); storeBt(0);
#ifdef HAVE_ASYNC_LDS
    __builtin_amdgcn_s_wait_asynccnt(0);
#endif
    __syncthreads();

    const int nK = K / BK;
    for (int k = 0; k < nK; ++k) {
        const int cur  = k & 1;
        const int nxt  = cur ^ 1;
        const bool more = (k + 1) < nK;

        // issue next-step global traffic before compute (latency overlap)
        if (more) {
#ifdef HAVE_ASYNC_LDS
            stageA_async((k + 1) * BK, nxt);
#else
            loadA((k + 1) * BK);
#endif
            loadB((k + 1) * BK);
        }

        // ---- compute on buffer `cur`: 8 WMMA per wave ---------------------
        v16h bf[2];
        #pragma unroll
        for (int nb = 0; nb < 2; ++nb) {
            const half_t* p = &Bt[cur][wn + nb * 16 + l16][lhalf * 16];
            bf[nb] = cat16(*(const v8h*)p, *(const v8h*)(p + 8));
        }
        #pragma unroll
        for (int mb = 0; mb < 4; ++mb) {
            const half_t* p = &As[cur][wm + mb * 16 + l16][lhalf * 16];
            v16h af = cat16(*(const v8h*)p, *(const v8h*)(p + 8));
            #pragma unroll
            for (int nb = 0; nb < 2; ++nb)
                acc[mb][nb] = wmma_f16(af, bf[nb], acc[mb][nb]);
        }

        // ---- drain next-step registers into buffer `nxt` ------------------
        if (more) {
            storeBt(nxt);
#ifndef HAVE_ASYNC_LDS
            storeA(nxt);
#endif
        }
#ifdef HAVE_ASYNC_LDS
        __builtin_amdgcn_s_wait_asynccnt(0);
#endif
        __syncthreads();
    }

    // ---- epilogue: bounce through LDS for coalesced 16B global stores -----
    half_t (*Cs)[BN + 8] = reinterpret_cast<half_t (*)[BN + 8]>(smem_raw);
    #pragma unroll
    for (int mb = 0; mb < 4; ++mb)
        #pragma unroll
        for (int nb = 0; nb < 2; ++nb)
            #pragma unroll
            for (int r = 0; r < 8; ++r)
                Cs[wm + mb * 16 + r + 8 * lhalf][wn + nb * 16 + l16] =
                    (half_t)acc[mb][nb][r];
    __syncthreads();
    #pragma unroll
    for (int p = 0; p < 8; ++p) {
        int idx = tid + p * 256;                // 0..2047
        int r   = idx >> 4;                     // 0..127
        int c   = (idx & 15) * 8;               // 0..120
        *(v8h*)(C + (size_t)(m0 + r) * N + n0 + c) = *(const v8h*)&Cs[r][c];
    }
}

// ===========================================================================
// Flash attention. grid = (SEQ/64, B*HEADS), block = 128 threads (4 waves).
// Each wave: 16 q-rows. KV chunk = 32. Q, K read from global row-major [S,64]
// (naturally matches A / B(=K^T) fragment layouts). V transposed into LDS.
// ===========================================================================
__global__ __launch_bounds__(128)
void attn_kernel(const half_t* __restrict__ Q,
                 const half_t* __restrict__ Kv,
                 const half_t* __restrict__ V,
                 half_t* __restrict__ O) {
    constexpr int HEADS = 32, KVH = 8, D = 64, SEQ = 2048;
    constexpr int GRP = HEADS / KVH;       // 4
    constexpr int KVC = 32;                // kv chunk
    constexpr int QSTRIDE = HEADS * D;     // 2048
    constexpr int KSTRIDE = KVH * D;       // 512

    const int bh   = blockIdx.y;
    const int b    = bh / HEADS;
    const int h    = bh % HEADS;
    const int kvh  = h / GRP;
    const int tid  = threadIdx.x;
    const int lane = tid & 31;
    const int wave = tid >> 5;
    const int l16  = lane & 15;
    const int lhalf = lane >> 4;

    const half_t* Qb = Q  + (size_t)b * SEQ * QSTRIDE + h   * D;
    const half_t* Kb = Kv + (size_t)b * SEQ * KSTRIDE + kvh * D;
    const half_t* Vb = V  + (size_t)b * SEQ * KSTRIDE + kvh * D;

    const int qrow = blockIdx.x * 64 + wave * 16 + l16;

    // Q fragments stay in registers for whole kernel (D=64 -> 2 k-steps)
    v16h qf[2];
    #pragma unroll
    for (int ks = 0; ks < 2; ++ks) {
        const half_t* p = Qb + (size_t)qrow * QSTRIDE + ks * 32 + lhalf * 16;
        qf[ks] = cat16(*(const v8h*)p, *(const v8h*)(p + 8));
    }

    __shared__ half_t Vt[D][KVC + 8];        // V chunk transposed: [d][kv]
    __shared__ half_t Pl[4][16][KVC + 8];    // per-wave probabilities

    v8f  o[4] = {};                          // 16 x 64 output accumulators
    float mrow[8], srow[8];
    #pragma unroll
    for (int r = 0; r < 8; ++r) { mrow[r] = -1e30f; srow[r] = 0.0f; }

    const float scale = 0.125f;              // 1/sqrt(64)

    for (int kc = 0; kc < SEQ; kc += KVC) {
        __syncthreads();                     // prior-iteration Vt reads done
        // ---- stage V chunk transposed (all 128 threads) -------------------
        #pragma unroll
        for (int p2 = 0; p2 < 2; ++p2) {
            int idx = tid + p2 * 128;        // 0..255
            int r   = idx >> 3;              // kv row 0..31
            int c   = (idx & 7) * 8;         // d offset
            v8h v = *(const v8h*)(Vb + (size_t)(kc + r) * KSTRIDE + c);
            #pragma unroll
            for (int j = 0; j < 8; ++j) Vt[c + j][r] = v[j];
        }

        // ---- S = Q K^T : 16 x 32, B fragment straight from global K -------
        v8f s[2] = {};
        #pragma unroll
        for (int nb = 0; nb < 2; ++nb)
            #pragma unroll
            for (int ks = 0; ks < 2; ++ks) {
                const half_t* p = Kb + (size_t)(kc + nb * 16 + l16) * KSTRIDE
                                     + ks * 32 + lhalf * 16;
                v16h kf = cat16(*(const v8h*)p, *(const v8h*)(p + 8));
                s[nb] = wmma_f16(qf[ks], kf, s[nb]);
            }

        // ---- online softmax (row = r + 8*lhalf; cols spread over 16 lanes)
        #pragma unroll
        for (int r = 0; r < 8; ++r) {
            float mx = fmaxf(s[0][r], s[1][r]);
            #pragma unroll
            for (int off = 1; off < 16; off <<= 1)
                mx = fmaxf(mx, __shfl_xor(mx, off, 32));
            mx *= scale;
            float mnew  = fmaxf(mrow[r], mx);
            float alpha = __expf(mrow[r] - mnew);
            mrow[r] = mnew;

            float p0 = __expf(s[0][r] * scale - mnew);
            float p1 = __expf(s[1][r] * scale - mnew);
            float rs = p0 + p1;
            #pragma unroll
            for (int off = 1; off < 16; off <<= 1)
                rs += __shfl_xor(rs, off, 32);
            srow[r] = srow[r] * alpha + rs;

            Pl[wave][r + 8 * lhalf][l16]      = (half_t)p0;
            Pl[wave][r + 8 * lhalf][16 + l16] = (half_t)p1;
            #pragma unroll
            for (int nb = 0; nb < 4; ++nb) o[nb][r] *= alpha;
        }
        __syncthreads();                     // Vt staged (P is same-wave)

        // ---- O += P @ V ---------------------------------------------------
        const half_t* pp = &Pl[wave][l16][lhalf * 16];
        v16h pf = cat16(*(const v8h*)pp, *(const v8h*)(pp + 8));
        #pragma unroll
        for (int nb = 0; nb < 4; ++nb) {
            const half_t* vp = &Vt[nb * 16 + l16][lhalf * 16];
            v16h vf = cat16(*(const v8h*)vp, *(const v8h*)(vp + 8));
            o[nb] = wmma_f16(pf, vf, o[nb]);
        }
    }

    // ---- epilogue: normalize, store f16 to O[b*SEQ + row][h*64 + col] ----
    #pragma unroll
    for (int nb = 0; nb < 4; ++nb)
        #pragma unroll
        for (int r = 0; r < 8; ++r) {
            int row = blockIdx.x * 64 + wave * 16 + r + 8 * lhalf;
            int col = nb * 16 + l16;
            float val = o[nb][r] / srow[r];
            O[(size_t)(b * SEQ + row) * QSTRIDE + h * D + col] = (half_t)val;
        }
}

// ===========================================================================
extern "C" void kernel_launch(void* const* d_in, const int* in_sizes, int n_in,
                              void* d_out, int out_size, void* d_ws, size_t ws_size,
                              hipStream_t stream) {
    const half_t* x  = (const half_t*)d_in[0];
    const half_t* Wq = (const half_t*)d_in[1];
    const half_t* Wk = (const half_t*)d_in[2];
    const half_t* Wv = (const half_t*)d_in[3];
    const half_t* Wo = (const half_t*)d_in[4];
    half_t* out = (half_t*)d_out;

    const int M = 2 * 2048;                  // 4096 token rows
    half_t* Qb = (half_t*)d_ws;              // [4096, 2048]
    half_t* Kb = Qb + (size_t)M * 2048;      // [4096, 512]
    half_t* Vb = Kb + (size_t)M * 512;       // [4096, 512]
    half_t* Ob = Vb + (size_t)M * 512;       // [4096, 2048]

    dim3 blk(256);
    // QKV projections (K = 2048 for all)
    gemm_f16_kernel<<<dim3(2048 / 128, M / 128), blk, 0, stream>>>(x, Wq, Qb, M, 2048, 2048);
    gemm_f16_kernel<<<dim3(512  / 128, M / 128), blk, 0, stream>>>(x, Wk, Kb, M, 512,  2048);
    gemm_f16_kernel<<<dim3(512  / 128, M / 128), blk, 0, stream>>>(x, Wv, Vb, M, 512,  2048);
    // flash attention: 32 q-tiles x 64 batch-heads
    attn_kernel<<<dim3(2048 / 64, 2 * 32), dim3(128), 0, stream>>>(Qb, Kb, Vb, Ob);
    // output projection
    gemm_f16_kernel<<<dim3(2048 / 128, M / 128), blk, 0, stream>>>(Ob, Wo, out, M, 2048, 2048);
}
